// FeatureExtractor_11149735101031
// MI455X (gfx1250) — matrix operand order repaired and checked by reading, not compiled, not run on hardware
//
#include <hip/hip_runtime.h>

#define BB 32
#define TT 50
#define NN 1024
#define CC 128
#define HH 64
#define GG 192  // 3*H

typedef __attribute__((ext_vector_type(16))) __bf16 v16bf;
typedef __attribute__((ext_vector_type(8)))  __bf16 v8bf;
typedef __attribute__((ext_vector_type(8)))  float  v8f;
typedef __attribute__((ext_vector_type(4)))  float  v4f;
typedef __attribute__((ext_vector_type(4)))  int    v4i;

// LDS layout (bf16 weight tiles, padded row strides to avoid bank conflicts;
// strides kept multiples of 8 elements so each lane's 16B chunk stays b128-aligned)
#define WIH_STRIDE 136   // 128 cols + 8 pad (272B rows)
#define WHH_STRIDE 72    // 64 cols + 8 pad  (144B rows)
#define WIH_ELEMS  (GG * WIH_STRIDE)                 // 26112
#define WHH_ELEMS  (GG * WHH_STRIDE)                 // 13824
#define BF16_BYTES ((WIH_ELEMS + WHH_ELEMS) * 2)     // 79872
#define BIAS_RZ_B  (BF16_BYTES)                      // 128 floats: b_ih+b_hh for r,z
#define BIAS_IN_B  (BIAS_RZ_B + 128 * 4)             // 64 floats: b_ih (n gate)
#define BIAS_HN_B  (BIAS_IN_B + 64 * 4)              // 64 floats: b_hh (n gate)
#define LDS_TOTAL  (BIAS_HN_B + 64 * 4)              // 80896 bytes

// Scheduling fence (emits no ISA): stops the pre-RA scheduler from hoisting
// whole GEMM passes worth of loads and blowing past 256 VGPRs into scratch.
__device__ __forceinline__ void sfence() {
#if __has_builtin(__builtin_amdgcn_sched_barrier)
  __builtin_amdgcn_sched_barrier(0);
#endif
}

// Scheduling group pin: mask 0x100 = DS-read, 0x8 = MFMA/WMMA.
// Builtin demands integer-constant args -> template parameters.
template <int MASK, int SIZE>
__device__ __forceinline__ void sgb() {
#if __has_builtin(__builtin_amdgcn_sched_group_barrier)
  __builtin_amdgcn_sched_group_barrier(MASK, SIZE, 0);
#endif
}

// Pipeline pattern for one 12-tile GEMM pass (24 ds_load_b128 + 12 wmma):
// prefetch 2 fragments, then {wmma, 2 ds loads} x10, then drain 2 wmmas.
// Keeps 2-3 fragments (~24 VGPRs) in flight so DS latency hides behind WMMA.
__device__ __forceinline__ void pipe12() {
  sgb<0x100, 4>();
#pragma unroll
  for (int i = 0; i < 10; ++i) { sgb<0x8, 1>(); sgb<0x100, 2>(); }
  sgb<0x8, 2>();
}

__device__ __forceinline__ float fast_tanh(float x) {
#if __has_builtin(__builtin_amdgcn_tanhf)
  return __builtin_amdgcn_tanhf(x);
#else
  float e = exp2f(x * -2.885390082f);
  return (1.0f - e) / (1.0f + e);
#endif
}
__device__ __forceinline__ float fast_sigmoid(float x) {
  return 0.5f + 0.5f * fast_tanh(0.5f * x);
}

// A-operand fragment (16x32 bf16) from padded row-major LDS weights.
// A layout used here (B uses the same k order): lane l holds row M=l%16; K = (l/16)*16 + e.
__device__ __forceinline__ v16bf ldWfrag(const __bf16* W, int stride, int grow,
                                         int kc, int hf) {
  const __bf16* p = W + grow * stride + kc + hf * 16;
  v8bf lo = *(const v8bf*)p;          // K_local = hf*16 + 0..7
  v8bf hi = *(const v8bf*)(p + 8);    // K_local = hf*16 + 8..15
  return __builtin_shufflevector(lo, hi, 0, 1, 2, 3, 4, 5, 6, 7,
                                 8, 9, 10, 11, 12, 13, 14, 15);
}

__device__ __forceinline__ v8f wmma_bf16(v16bf a, v16bf b, v8f c) {
  return __builtin_amdgcn_wmma_f32_16x16x32_bf16(false, a, false, b, (short)0, c,
                                                 false, false);
}

__global__ void __launch_bounds__(256)
gru_fused_wmma_kernel(const float* __restrict__ chars,
                      const float* __restrict__ W_ih,
                      const float* __restrict__ W_hh,
                      const float* __restrict__ b_ih,
                      const float* __restrict__ b_hh,
                      float* __restrict__ out) {
  extern __shared__ __align__(16) char smem[];
  __bf16* sWih = (__bf16*)smem;
  __bf16* sWhh = sWih + WIH_ELEMS;
  float* sBrz = (float*)(smem + BIAS_RZ_B);
  float* sBin = (float*)(smem + BIAS_IN_B);
  float* sBhn = (float*)(smem + BIAS_HN_B);

  const int tid = threadIdx.x;

  // ---- stage weights (fp32 -> bf16) + bias tables into LDS ----
  for (int i = tid; i < GG * CC; i += 256)
    sWih[(i >> 7) * WIH_STRIDE + (i & 127)] = (__bf16)W_ih[i];
  for (int i = tid; i < GG * HH; i += 256)
    sWhh[(i >> 6) * WHH_STRIDE + (i & 63)] = (__bf16)W_hh[i];
  if (tid < 128)       sBrz[tid] = b_ih[tid] + b_hh[tid];
  else if (tid < 192)  sBin[tid - 128] = b_ih[tid];          // b_ih[128..191]
  else                 sBhn[tid - 192] = b_hh[tid - 64];     // b_hh[128..191]
  __syncthreads();

  const int lane = tid & 31;
  const int hf   = lane >> 4;   // lane half (0/1)
  const int lr   = lane & 15;   // sequence (m) for B/C/D; gate row for A
  const int wid  = tid >> 5;

  const int rowbase = blockIdx.x * 128 + wid * 16;  // 16 sequences per wave
  const int bb = rowbase >> 10;
  const int n0 = rowbase & (NN - 1);

  // x^T B-fragment base: lane holds 16 consecutive channels of its row.
  const float* xlane =
      chars + ((size_t)(bb * TT) * NN + (size_t)(n0 + lr)) * CC + hf * 16;
  const size_t tstride = (size_t)NN * CC;

  v8f hreg[4];   // h in transposed D layout: col = j*16 + hf*8 + v, row m = lr
  v8bf hpk[4];   // bf16-packed h for next step's B operand
#pragma unroll
  for (int j = 0; j < 4; ++j)
#pragma unroll
    for (int v = 0; v < 8; ++v) { hreg[j][v] = 0.0f; hpk[j][v] = (__bf16)0.0f; }

  for (int t = 0; t < TT; ++t) {
    sfence();
    const float* xt = xlane + (size_t)t * tstride;

    // prefetch next timestep's 16x128 fp32 tile (512B/row, 2x128B per lane-half)
    {
      int tn = (t + 1 < TT) ? t + 1 : t;
      const float* pf = (xlane - hf * 16) + (size_t)tn * tstride + hf * 64;
      __builtin_prefetch(pf, 0, 3);
      __builtin_prefetch(pf + 32, 0, 3);
    }

    // ---- init accumulators with biases (folds b_ih/b_hh into the GEMMs) ----
    v8f acc_rz[8], acc_in[4], acc_hn[4];
#pragma unroll
    for (int j = 0; j < 8; ++j) acc_rz[j] = *(const v8f*)(sBrz + j * 16 + hf * 8);
#pragma unroll
    for (int j = 0; j < 4; ++j) acc_in[j] = *(const v8f*)(sBin + j * 16 + hf * 8);
#pragma unroll
    for (int j = 0; j < 4; ++j) acc_hn[j] = *(const v8f*)(sBhn + j * 16 + hf * 8);
    sfence();

    // ---- gi^T += W_ih * x^T  (K = 128, 4 k-steps x 12 gate tiles) ----
#pragma unroll
    for (int ks = 0; ks < 4; ++ks) {
      const float* xp = xt + ks * 32;
      v4f f0 = *(const v4f*)(xp);
      v4f f1 = *(const v4f*)(xp + 4);
      v4f f2 = *(const v4f*)(xp + 8);
      v4f f3 = *(const v4f*)(xp + 12);
      v16bf xb;
#pragma unroll
      for (int e = 0; e < 4; ++e) {
        xb[e]      = (__bf16)f0[e];
        xb[4 + e]  = (__bf16)f1[e];
        xb[8 + e]  = (__bf16)f2[e];
        xb[12 + e] = (__bf16)f3[e];
      }
      sfence();  // f32 x values die here; only xb (8 VGPRs) stays live
#pragma unroll
      for (int jt = 0; jt < 12; ++jt) {
        v16bf wa = ldWfrag(sWih, WIH_STRIDE, jt * 16 + lr, ks * 32, hf);
        if (jt < 8)
          acc_rz[jt] = wmma_bf16(wa, xb, acc_rz[jt]);
        else
          acc_in[jt - 8] = wmma_bf16(wa, xb, acc_in[jt - 8]);
      }
      pipe12();  // pin ds/wmma software pipeline for this pass
      sfence();
    }

    // ---- gh^T += W_hh * h^T  (h rebuilt from registers via lane-half swap) ----
    v4i hw[4], ow[4];
#pragma unroll
    for (int j = 0; j < 4; ++j) {
      hw[j] = __builtin_bit_cast(v4i, hpk[j]);
      v4i s;
#pragma unroll
      for (int k = 0; k < 4; ++k) s[k] = __shfl_xor(hw[j][k], 16, 32);
      ow[j] = s;
    }
    v16bf hbf[2];
#pragma unroll
    for (int ks = 0; ks < 2; ++ks) {
      // lane needs h cols ks*32 + hf*16 + 0..15 at m=lr:
      // low 8 from tile (2ks+hf) sub 0..7, high 8 from sub 8..15 (other half)
      v4i lo = hf ? ow[2 * ks + 1] : hw[2 * ks];
      v4i hi = hf ? hw[2 * ks + 1] : ow[2 * ks];
      v8bf lob = __builtin_bit_cast(v8bf, lo);
      v8bf hib = __builtin_bit_cast(v8bf, hi);
      hbf[ks] = __builtin_shufflevector(lob, hib, 0, 1, 2, 3, 4, 5, 6, 7,
                                        8, 9, 10, 11, 12, 13, 14, 15);
    }
    sfence();
#pragma unroll
    for (int ks = 0; ks < 2; ++ks) {
#pragma unroll
      for (int jt = 0; jt < 12; ++jt) {
        v16bf wa = ldWfrag(sWhh, WHH_STRIDE, jt * 16 + lr, ks * 32, hf);
        if (jt < 8)
          acc_rz[jt] = wmma_bf16(wa, hbf[ks], acc_rz[jt]);
        else
          acc_hn[jt - 8] = wmma_bf16(wa, hbf[ks], acc_hn[jt - 8]);
      }
      pipe12();
      sfence();
    }

    // ---- gates: r,z sigmoid; n = tanh(i_n + r*h_n); h = (1-z)*n + z*h ----
#pragma unroll
    for (int j = 0; j < 4; ++j) {
#pragma unroll
      for (int v = 0; v < 8; ++v) {
        float r = fast_sigmoid(acc_rz[j][v]);
        float z = fast_sigmoid(acc_rz[j + 4][v]);
        float nn = fast_tanh(acc_in[j][v] + r * acc_hn[j][v]);
        float h = hreg[j][v];
        h = nn + z * (h - nn);
        hreg[j][v] = h;
        hpk[j][v] = (__bf16)h;
      }
    }
    sfence();
  }

  // ---- store h_final: out[(b*N+n)*H + col], col = j*16 + hf*8 + v ----
  float* op = out + (size_t)(rowbase + lr) * HH;
#pragma unroll
  for (int j = 0; j < 4; ++j) {
    v4f a, b2;
#pragma unroll
    for (int v = 0; v < 4; ++v) { a[v] = hreg[j][v]; b2[v] = hreg[j][v + 4]; }
    *(v4f*)(op + j * 16 + hf * 8) = a;
    *(v4f*)(op + j * 16 + hf * 8 + 4) = b2;
  }
}

extern "C" void kernel_launch(void* const* d_in, const int* in_sizes, int n_in,
                              void* d_out, int out_size, void* d_ws, size_t ws_size,
                              hipStream_t stream) {
  (void)in_sizes; (void)n_in; (void)out_size; (void)d_ws; (void)ws_size;
  const float* chars = (const float*)d_in[0];
  const float* W_ih  = (const float*)d_in[1];
  const float* W_hh  = (const float*)d_in[2];
  const float* b_ih  = (const float*)d_in[3];
  const float* b_hh  = (const float*)d_in[4];
  float* out = (float*)d_out;

  // 80,896B dynamic LDS (gfx1250 WGP has 320KB)
  (void)hipFuncSetAttribute((const void*)gru_fused_wmma_kernel,
                            hipFuncAttributeMaxDynamicSharedMemorySize, LDS_TOTAL);

  // 32768 sequences / (8 waves * 16 rows) = 256 blocks of 256 threads
  gru_fused_wmma_kernel<<<256, 256, LDS_TOTAL, stream>>>(chars, W_ih, W_hh,
                                                         b_ih, b_hh, out);
}
